// SelfAttention_34737695490256
// MI455X (gfx1250) — matrix operand order repaired and checked by reading, not compiled, not run on hardware
//
#include <hip/hip_runtime.h>
#include <hip/hip_bf16.h>

// ---------------- problem constants (from reference) ----------------
#define BB 2
#define SS 2048
#define DD 1024
#define HH 16
#define HD 64
#define LOG2E 1.44269504088896340736f

typedef __attribute__((ext_vector_type(8)))  __bf16 v8bf;
typedef __attribute__((ext_vector_type(16))) __bf16 v16bf;
typedef __attribute__((ext_vector_type(8)))  float  v8f;

__device__ __forceinline__ unsigned short f2bf_u(float f) {
    unsigned int u = __float_as_uint(f);
    u += 0x7FFFu + ((u >> 16) & 1u);   // round-to-nearest-even
    return (unsigned short)(u >> 16);
}

__device__ __forceinline__ float fast_exp2(float x) {
    return __builtin_amdgcn_exp2f(x);   // v_exp_f32
}

__device__ __forceinline__ v16bf cat16(v8bf lo, v8bf hi) {
    return __builtin_shufflevector(lo, hi, 0,1,2,3,4,5,6,7,8,9,10,11,12,13,14,15);
}

__device__ __forceinline__ v16bf load_frag16(const unsigned short* p0, const unsigned short* p1) {
    v8bf lo = *reinterpret_cast<const v8bf*>(p0);
    v8bf hi = *reinterpret_cast<const v8bf*>(p1);
    return cat16(lo, hi);
}

__device__ __forceinline__ v8f wmma_bf16(v16bf a, v16bf b, v8f c) {
    return __builtin_amdgcn_wmma_f32_16x16x32_bf16(false, a, false, b, (short)0, c, false, false);
}

// ---------------- stage 0: fp32 -> bf16 ----------------
__global__ void cvt_f32_bf16(const float* __restrict__ in, unsigned short* __restrict__ out, int n) {
    int i = blockIdx.x * blockDim.x + threadIdx.x;
    int stride = gridDim.x * blockDim.x;
    for (; i < n; i += stride) out[i] = f2bf_u(in[i]);
}

// ---------------- stage 1: fused QKV projection (bf16 WMMA) ----------------
// y[m,e] = sum_k x[m,k] * W[e,k] + bias[e]   (nn.Linear: x @ W^T + b)
// Each wave: 32(M) x 64(N) tile of one of {Q,K,V}; K-dim double-buffered (ping-pong)
// so next-chunk loads are in flight across current-chunk WMMAs.
// Q written pre-scaled by log2(e)/sqrt(HD) to [b,h,s,hd]; K to [b,h,s,hd]; V transposed to [b,h,hd,s].
__global__ void __launch_bounds__(256) qkv_kernel(
    const unsigned short* __restrict__ X,   // [B*S, D] bf16
    const unsigned short* __restrict__ Wq,
    const unsigned short* __restrict__ Wk,
    const unsigned short* __restrict__ Wv,
    const float* __restrict__ bq,
    const float* __restrict__ bk,
    const float* __restrict__ bv,
    unsigned short* __restrict__ Qo,        // [B,H,S,HD]
    unsigned short* __restrict__ Ko,        // [B,H,S,HD]
    unsigned short* __restrict__ Vt)        // [B,H,HD,S]
{
    const int wid  = threadIdx.x >> 5;
    const int lane = threadIdx.x & 31;
    const int half = lane >> 4;
    const int l15  = lane & 15;

    const int job   = blockIdx.x * (blockDim.x >> 5) + wid;
    const int ntile = job & 15;            // 16 tiles of 64 output features
    const int mtile = (job >> 4) & 127;    // 128 tiles of 32 rows (B*S/32)
    const int mat   = job >> 11;           // 0=Q 1=K 2=V
    if (mat >= 3) return;

    const unsigned short* W    = (mat == 0) ? Wq : (mat == 1) ? Wk : Wv;
    const float*          bias = (mat == 0) ? bq : (mat == 1) ? bk : bv;

    const int m0 = mtile * 32;
    const int n0 = ntile * 64;

    v8f acc[2][4] = {};

    // fragment loaders --------------------------------------------------
    auto loadA = [&](int k0, v16bf a[2]) {
#pragma unroll
        for (int mt = 0; mt < 2; ++mt) {
            const unsigned int ab = (unsigned int)(m0 + 16 * mt + l15) * DD + k0 + half * 8;
            a[mt] = load_frag16(X + ab, X + ab + 16);
        }
    };
    auto loadB = [&](int k0, v16bf bf[4]) {
#pragma unroll
        for (int t = 0; t < 4; ++t) {
            const unsigned int bb = (unsigned int)(n0 + 16 * t + l15) * DD + k0 + half * 8;
            bf[t] = load_frag16(W + bb, W + bb + 16);
        }
    };

    v16bf aA[2], bA[4], aB[2], bB[4];
    loadA(0, aA);
    loadB(0, bA);

    for (int k0 = 0; k0 < DD; k0 += 64) {
        // prefetch chunk k0+32 while computing chunk k0
        loadA(k0 + 32, aB);
        loadB(k0 + 32, bB);
#pragma unroll
        for (int mt = 0; mt < 2; ++mt)
#pragma unroll
            for (int t = 0; t < 4; ++t)
                acc[mt][t] = wmma_bf16(aA[mt], bA[t], acc[mt][t]);

        // prefetch chunk k0+64 (last one overshoots harmlessly inside ws)
        loadA(k0 + 64, aA);
        loadB(k0 + 64, bA);
#pragma unroll
        for (int mt = 0; mt < 2; ++mt)
#pragma unroll
            for (int t = 0; t < 4; ++t)
                acc[mt][t] = wmma_bf16(aB[mt], bB[t], acc[mt][t]);
    }

    // Epilogue: C-layout (VGPR r, lane<16 -> M=r ; lane>=16 -> M=r+8), N = n0+16t+l15
#pragma unroll
    for (int mt = 0; mt < 2; ++mt) {
#pragma unroll
        for (int t = 0; t < 4; ++t) {
            const int e  = n0 + 16 * t + l15;
            const int h  = e >> 6;
            const int hd = e & 63;
            const float bs = bias[e];
#pragma unroll
            for (int r = 0; r < 8; ++r) {
                const int m = m0 + 16 * mt + r + half * 8;  // flat over B*S
                const int b = m >> 11;                      // /S
                const int s = m & (SS - 1);
                const float y = acc[mt][t][r] + bs;
                if (mat == 0) {
                    // fold 1/sqrt(HD) and the exp->exp2 conversion into Q
                    Qo[(((unsigned)(b * HH + h) * SS) + s) * HD + hd] = f2bf_u(y * (0.125f * LOG2E));
                } else if (mat == 1) {
                    Ko[(((unsigned)(b * HH + h) * SS) + s) * HD + hd] = f2bf_u(y);
                } else {
                    Vt[(((unsigned)(b * HH + h) * HD) + hd) * SS + s] = f2bf_u(y);
                }
            }
        }
    }
}

// ---------------- stage 2: flash attention (bf16 WMMA, online softmax in exp2 domain) ----------------
// One wave per (b, h, 32-row q block). KV processed in 32-column chunks.
// K and V fragments are shared by both 16-row q tiles; V loads issued early to
// overlap the score WMMAs and softmax VALU work.
__global__ void __launch_bounds__(256) attn_kernel(
    const unsigned short* __restrict__ Q,    // [B,H,S,HD], pre-scaled by 0.125*log2(e)
    const unsigned short* __restrict__ K,    // [B,H,S,HD]
    const unsigned short* __restrict__ Vt,   // [B,H,HD,S]
    const float* __restrict__ mask,          // [B,S] additive (pre exp2-domain)
    float* __restrict__ out)                 // [B,S,D]
{
    __shared__ __attribute__((aligned(16))) unsigned short plds[8 * 2 * 16 * 32]; // per wave: two 16x32 bf16 P tiles

    const int wid  = threadIdx.x >> 5;
    const int lane = threadIdx.x & 31;
    const int half = lane >> 4;
    const int l15  = lane & 15;

    const int job   = blockIdx.x * (blockDim.x >> 5) + wid;
    const int qblk  = job & 63;          // 64 blocks of 32 q rows
    const int h     = (job >> 6) & 15;
    const int b     = job >> 10;
    const int q0    = qblk << 5;

    const unsigned long headQK = (unsigned long)(b * HH + h) * (SS * HD);
    const unsigned long headV  = (unsigned long)(b * HH + h) * (HD * SS);

    // Q A-fragments: [q tile][K-dim chunk]
    v16bf qa[2][2];
#pragma unroll
    for (int qt = 0; qt < 2; ++qt) {
        const unsigned long base = headQK + (unsigned long)(q0 + 16 * qt + l15) * HD;
#pragma unroll
        for (int c = 0; c < 2; ++c) {
            const unsigned long ab = base + 32 * c + half * 8;
            qa[qt][c] = load_frag16(Q + ab, Q + ab + 16);
        }
    }

    v8f o[2][4] = {};
    float mrow[2][8], lrow[2][8];
#pragma unroll
    for (int qt = 0; qt < 2; ++qt)
#pragma unroll
        for (int r = 0; r < 8; ++r) { mrow[qt][r] = -3.0e38f; lrow[qt][r] = 0.0f; }

    unsigned short* myP = plds + wid * (2 * 16 * 32);

    for (int kc = 0; kc < SS; kc += 32) {
        // V B-fragments first: overlap with score WMMAs + softmax
        v16bf vf[4];
#pragma unroll
        for (int t = 0; t < 4; ++t) {
            const unsigned long vb = headV + (unsigned long)(16 * t + l15) * SS + kc + half * 8;
            vf[t] = load_frag16(Vt + vb, Vt + vb + 16);
        }

        // K^T B-fragments: [kv 16-col tile][K-dim chunk]
        v16bf kf[2][2];
#pragma unroll
        for (int n = 0; n < 2; ++n)
#pragma unroll
            for (int c = 0; c < 2; ++c) {
                const unsigned long kb = headQK + (unsigned long)(kc + 16 * n + l15) * HD + 32 * c + half * 8;
                kf[n][c] = load_frag16(K + kb, K + kb + 16);
            }

        // scores: [q tile][kv tile]
        v8f sc[2][2] = {};
#pragma unroll
        for (int c = 0; c < 2; ++c)
#pragma unroll
            for (int qt = 0; qt < 2; ++qt)
#pragma unroll
                for (int n = 0; n < 2; ++n)
                    sc[qt][n] = wmma_bf16(qa[qt][c], kf[n][c], sc[qt][n]);

        const float mk0 = mask[b * SS + kc + l15] * LOG2E;
        const float mk1 = mask[b * SS + kc + 16 + l15] * LOG2E;

        // online softmax per row (exp2 domain); rows live across 16-lane halves
#pragma unroll
        for (int qt = 0; qt < 2; ++qt) {
#pragma unroll
            for (int r = 0; r < 8; ++r) {
                float a0 = sc[qt][0][r] + mk0;
                float a1 = sc[qt][1][r] + mk1;
                float mx = fmaxf(a0, a1);
                mx = fmaxf(mx, __shfl_xor(mx, 1));
                mx = fmaxf(mx, __shfl_xor(mx, 2));
                mx = fmaxf(mx, __shfl_xor(mx, 4));
                mx = fmaxf(mx, __shfl_xor(mx, 8));
                const float mnew = fmaxf(mrow[qt][r], mx);
                const float corr = fast_exp2(mrow[qt][r] - mnew);
                const float p0 = fast_exp2(a0 - mnew);
                const float p1 = fast_exp2(a1 - mnew);
                float sum = p0 + p1;
                sum += __shfl_xor(sum, 1);
                sum += __shfl_xor(sum, 2);
                sum += __shfl_xor(sum, 4);
                sum += __shfl_xor(sum, 8);
                lrow[qt][r] = lrow[qt][r] * corr + sum;
                mrow[qt][r] = mnew;
#pragma unroll
                for (int t = 0; t < 4; ++t) o[qt][t][r] *= corr;

                const int prow = r + half * 8;
                myP[qt * 512 + prow * 32 + l15]      = f2bf_u(p0);
                myP[qt * 512 + prow * 32 + 16 + l15] = f2bf_u(p1);
            }
        }

        // cross-lane LDS RAW within the wave: DS pipe is in-order; fence compiler + pipeline
        asm volatile("s_wait_dscnt 0x0" ::: "memory");

        // reload P tiles (16x32) as A-fragments
        v16bf pf[2];
#pragma unroll
        for (int qt = 0; qt < 2; ++qt) {
            const int off = qt * 512 + l15 * 32 + half * 8;
            pf[qt] = load_frag16(myP + off, myP + off + 16);
        }

        // P @ V : 4 head-dim tiles of 16 columns, K=32 kv positions, shared V frags
#pragma unroll
        for (int qt = 0; qt < 2; ++qt)
#pragma unroll
            for (int t = 0; t < 4; ++t)
                o[qt][t] = wmma_bf16(pf[qt], vf[t], o[qt][t]);
    }

    // epilogue: normalize by l and store fp32 ctx [b,s, h*64+hd]
#pragma unroll
    for (int qt = 0; qt < 2; ++qt) {
#pragma unroll
        for (int r = 0; r < 8; ++r) {
            const float inv = 1.0f / lrow[qt][r];
            const int s = q0 + 16 * qt + r + half * 8;
            const unsigned long ob = ((unsigned long)(b * SS + s)) * DD + h * HD;
#pragma unroll
            for (int t = 0; t < 4; ++t)
                out[ob + 16 * t + l15] = o[qt][t][r] * inv;
        }
    }
}

// ---------------- host side ----------------
extern "C" void kernel_launch(void* const* d_in, const int* in_sizes, int n_in,
                              void* d_out, int out_size, void* d_ws, size_t ws_size,
                              hipStream_t stream) {
    const float* hs   = (const float*)d_in[0];  // [B,S,D]
    const float* mask = (const float*)d_in[1];  // [B,1,1,S]
    const float* Wq   = (const float*)d_in[2];
    const float* bq   = (const float*)d_in[3];
    const float* Wk   = (const float*)d_in[4];
    const float* bk   = (const float*)d_in[5];
    const float* Wv   = (const float*)d_in[6];
    const float* bv   = (const float*)d_in[7];
    float* out = (float*)d_out;

    const size_t NX = (size_t)BB * SS * DD;  // 4194304
    const size_t NW = (size_t)DD * DD;       // 1048576

    unsigned short* ws  = (unsigned short*)d_ws;
    unsigned short* Xbf = ws;
    unsigned short* Wqb = Xbf + NX;
    unsigned short* Wkb = Wqb + NW;
    unsigned short* Wvb = Wkb + NW;
    unsigned short* Qb  = Wvb + NW;
    unsigned short* Kb  = Qb + NX;
    unsigned short* Vtb = Kb + NX;           // total ~38 MiB of ws

    cvt_f32_bf16<<<4096, 256, 0, stream>>>(hs, Xbf, (int)NX);
    cvt_f32_bf16<<<1024, 256, 0, stream>>>(Wq, Wqb, (int)NW);
    cvt_f32_bf16<<<1024, 256, 0, stream>>>(Wk, Wkb, (int)NW);
    cvt_f32_bf16<<<1024, 256, 0, stream>>>(Wv, Wvb, (int)NW);

    // jobs = 3 mats * 128 m-tiles * 16 n64-tiles = 6144 waves, 8 waves/block
    qkv_kernel<<<768, 256, 0, stream>>>(Xbf, Wqb, Wkb, Wvb, bq, bk, bv, Qb, Kb, Vtb);

    // jobs = B*H*(S/32) = 2048 waves, 8 waves/block
    attn_kernel<<<256, 256, 0, stream>>>(Qb, Kb, Vtb, mask, out);
}
